// MutuallyRecursiveRNN_47931835024082
// MI455X (gfx1250) — compile-verified, matrix operand order checked by reading, not run on hardware
//
#include <hip/hip_runtime.h>
#include <hip/hip_bf16.h>
#include <stdint.h>
#include <stddef.h>

typedef _Float16 v8h  __attribute__((ext_vector_type(8)));
typedef _Float16 v16h __attribute__((ext_vector_type(16)));
typedef float    v8f  __attribute__((ext_vector_type(8)));

#define B_   512
#define S_   128
#define F_   512
#define E_   512            // EP == EM
#define KD   1536           // F_ + E_ (Wih part) + E_ (Whh part)
#define NG   2048           // 4*E gates
#define NKT  (KD/32)        // 48 K-fragments of 32
#define NNT  (NG/16)        // 128 N-tiles of 16
#define DEPTH 4             // async LDS pipeline stages (NKT % DEPTH == 0)

// ---------------------------------------------------------------------------
// Pack [Wih(n,0..1023) ; Whh(n,0..511)] -> fp16 WMMA B fragments.
// Fragment (nt,kt) stored lane-major: 32 lanes x 16 halfs (32B/lane).
// ---------------------------------------------------------------------------
__global__ __launch_bounds__(256) void pack_b_kernel(
    const float* __restrict__ Wih, const float* __restrict__ Whh,
    _Float16* __restrict__ Bp)
{
    int gid  = blockIdx.x * 256 + threadIdx.x;
    int fid  = gid >> 5;
    int lane = gid & 31;
    if (fid >= NNT * NKT) return;
    int nt   = fid / NKT;
    int kt   = fid % NKT;
    int n    = nt * 16 + (lane & 15);
    int half = lane >> 4;

    v8h lo, hi;
    #pragma unroll
    for (int v = 0; v < 8; ++v) {
        int kb = ((v < 4) ? 2 * v : 16 + 2 * (v - 4)) + 8 * half;
        #pragma unroll
        for (int p = 0; p < 2; ++p) {
            int k = kt * 32 + kb + p;
            float w = (k < F_ + E_) ? Wih[(size_t)n * (F_ + E_) + k]
                                    : Whh[(size_t)n * E_ + (k - (F_ + E_))];
            if (v < 4) lo[2 * v + p] = (_Float16)w;
            else       hi[2 * (v - 4) + p] = (_Float16)w;
        }
    }
    _Float16* dst = Bp + (size_t)fid * 512 + lane * 16;
    *(v8h*)(dst)     = lo;
    *(v8h*)(dst + 8) = hi;
}

// bias[n] = bih[n] + bhh[n] for both cells
__global__ __launch_bounds__(256) void bias_init_kernel(
    const float* __restrict__ mbih, const float* __restrict__ mbhh,
    const float* __restrict__ pbih, const float* __restrict__ pbhh,
    float* __restrict__ biasM, float* __restrict__ biasP)
{
    int i = blockIdx.x * 256 + threadIdx.x;
    if (i < NG)            biasM[i] = mbih[i] + mbhh[i];
    else if (i < 2 * NG) { int j = i - NG; biasP[j] = pbih[j] + pbhh[j]; }
}

// zero actM h-slots, phStage, and both cell states
__global__ __launch_bounds__(256) void init_state_kernel(
    _Float16* __restrict__ actM, _Float16* __restrict__ phStage,
    float* __restrict__ mc, float* __restrict__ pc)
{
    int i = blockIdx.x * 256 + threadIdx.x;        // 512*1024 threads
    int m = i >> 10, j = i & 1023;
    actM[(size_t)m * KD + 512 + j] = (_Float16)0.f;
    if (i < B_ * E_) { phStage[i] = (_Float16)0.f; mc[i] = 0.f; pc[i] = 0.f; }
}

// x_0 (fp32) -> fp16 into slot [0:512) of actM
__global__ __launch_bounds__(256) void convert_x_kernel(
    const float* __restrict__ x, _Float16* __restrict__ actM)
{
    int i = blockIdx.x * 256 + threadIdx.x;        // 512*512 threads
    int m = i >> 9, j = i & 511;
    actM[(size_t)m * KD + j] = (_Float16)x[((size_t)m * S_) * F_ + j];
}

// one B fragment (32 lanes x 32B) -> LDS, two async b128 ops (offset 0 / 16
// applies to BOTH the LDS and global addresses, so one laddr covers both)
__device__ __forceinline__ void async_copy_frag(uint32_t laddr, const _Float16* gp)
{
    asm volatile(
        "global_load_async_to_lds_b128 %0, %1, off\n\t"
        "global_load_async_to_lds_b128 %0, %1, off offset:16"
        :: "v"(laddr), "v"((uint64_t)(uintptr_t)gp)
        : "memory");
}

// ---------------------------------------------------------------------------
// Fully fused LSTM cell step:
//   gates = act_read[512x1536] * B + bias   (fp16 WMMA, f32 accum)
//   c,h update in-register; h written into the PARTNER act buffer + stage.
// Each wave owns a 16(M) x 16(j) tile of ALL FOUR gates (nt = gate*32 + jt),
// so after the k-loop lane holds i,f,g,o for 8 (m,j) elements. B-weights
// stream through a DEPTH=4 per-wave LDS ring via GLOBAL_LOAD_ASYNC_TO_LDS
// (issued 4 k-steps ahead); WMMAs only wait on ds_loads; no barriers.
// Kernel reads only act_read, writes only act_write -> no intra-kernel races:
//   mode 0 (metrics): act_write=[x copy | h_new | hStageIn], stage h_new
//   mode 1 (price):   act_write=[x_{tNext} | h_new | hStageIn], stage h_new
// ---------------------------------------------------------------------------
__global__ __launch_bounds__(128) void lstm_cell_kernel(
    const _Float16* __restrict__ act_read, const _Float16* __restrict__ Bp,
    const float* __restrict__ bias, float* __restrict__ c,
    _Float16* __restrict__ act_write,
    _Float16* __restrict__ hStageOut, const _Float16* __restrict__ hStageIn,
    float* __restrict__ h32,
    const float* __restrict__ x, int tNext, int mode)
{
    __shared__ _Float16 smem[4 * DEPTH * 2048];    // 64KB

    const int lane   = threadIdx.x & 31;
    const int wave   = threadIdx.x >> 5;           // 0..3
    const int mt     = blockIdx.x;                 // 0..31
    const int jt     = blockIdx.y * 4 + wave;      // 0..31
    const int row    = lane & 15;
    const int half   = lane >> 4;

    v8f acc[4];
    v8f zero = {0.f, 0.f, 0.f, 0.f, 0.f, 0.f, 0.f, 0.f};
    #pragma unroll
    for (int i = 0; i < 4; ++i) acc[i] = zero;

    // per-lane global bases: fragment column for gate i at tile (i*32 + jt)
    const _Float16* bb[4];
    #pragma unroll
    for (int i = 0; i < 4; ++i)
        bb[i] = Bp + ((size_t)(i * 32 + jt) * NKT) * 512 + lane * 16;

    // per-lane LDS base (flat-pointer low 32 bits == LDS byte offset)
    const uint32_t lwave =
        (uint32_t)(uintptr_t)(smem) + (uint32_t)wave * (DEPTH * 4096u)
        + (uint32_t)lane * 32u;
    const _Float16* swave = smem + wave * (DEPTH * 2048) + lane * 16;

    // A: register double buffer (16-bit A-fragment layout: two 16B runs/lane)
    const _Float16* arow = act_read + (size_t)(mt * 16 + row) * KD + half * 8;
    v8h a_lo = *(const v8h*)(arow);
    v8h a_hi = *(const v8h*)(arow + 16);

    // prologue: async-stage k-steps 0..DEPTH-1 (8 ops each, 32 in flight)
    #pragma unroll
    for (int d = 0; d < DEPTH; ++d) {
        uint32_t lk = lwave + ((uint32_t)d << 12);
        #pragma unroll
        for (int i = 0; i < 4; ++i)
            async_copy_frag(lk + (uint32_t)i * 1024u, bb[i] + (size_t)d * 512);
    }

    #pragma unroll 4
    for (int kt = 0; kt < NKT; ++kt) {
        const int d = kt & (DEPTH - 1);

        // stage d (k-step kt) ready when <= 24 newer async ops in flight
        asm volatile("s_wait_asynccnt 0x18" ::: "memory");

        const _Float16* sd = swave + d * 2048;
        v16h b0, b1, b2, b3;
        {
            v8h l0 = *(const v8h*)(sd +    0), h0 = *(const v8h*)(sd +    8);
            v8h l1 = *(const v8h*)(sd +  512), h1 = *(const v8h*)(sd +  520);
            v8h l2 = *(const v8h*)(sd + 1024), h2 = *(const v8h*)(sd + 1032);
            v8h l3 = *(const v8h*)(sd + 1536), h3 = *(const v8h*)(sd + 1544);
            #pragma unroll
            for (int e = 0; e < 8; ++e) {
                b0[e] = l0[e]; b0[8 + e] = h0[e];
                b1[e] = l1[e]; b1[8 + e] = h1[e];
                b2[e] = l2[e]; b2[8 + e] = h2[e];
                b3[e] = l3[e]; b3[8 + e] = h3[e];
            }
        }

        v16h a;
        #pragma unroll
        for (int e = 0; e < 8; ++e) { a[e] = a_lo[e]; a[8 + e] = a_hi[e]; }
        {
            const size_t kn = (kt + 1 < NKT) ? (size_t)(kt + 1) : (size_t)kt;
            a_lo = *(const v8h*)(arow + kn * 32);
            a_hi = *(const v8h*)(arow + kn * 32 + 16);
        }

        // drain DS reads of stage d, then refill with k-step kt+DEPTH (clamped)
        asm volatile("s_wait_dscnt 0x0" ::: "memory");
        {
            const size_t k4 = (kt + DEPTH < NKT) ? (size_t)(kt + DEPTH) : (size_t)kt;
            uint32_t lk = lwave + ((uint32_t)d << 12);
            #pragma unroll
            for (int i = 0; i < 4; ++i)
                async_copy_frag(lk + (uint32_t)i * 1024u, bb[i] + k4 * 512);
        }

        acc[0] = __builtin_amdgcn_wmma_f32_16x16x32_f16(false, a, false, b0, (short)0, acc[0], false, false);
        acc[1] = __builtin_amdgcn_wmma_f32_16x16x32_f16(false, a, false, b1, (short)0, acc[1], false, false);
        acc[2] = __builtin_amdgcn_wmma_f32_16x16x32_f16(false, a, false, b2, (short)0, acc[2], false, false);
        acc[3] = __builtin_amdgcn_wmma_f32_16x16x32_f16(false, a, false, b3, (short)0, acc[3], false, false);
    }

    // fused LSTM epilogue: lane element v -> (m = mt*16 + v + 8*half,
    // j = jt*16 + row); acc[0..3][v] = gates i,f,g,o pre-bias.
    const int j = jt * 16 + row;
    const float bI = bias[j],        bF = bias[512 + j];
    const float bG = bias[1024 + j], bO = bias[1536 + j];
    #pragma unroll
    for (int v = 0; v < 8; ++v) {
        const int m = mt * 16 + v + 8 * half;
        const size_t cm = (size_t)m * 512 + j;
        const size_t aw = (size_t)m * KD;
        float gi = acc[0][v] + bI;
        float gf = acc[1][v] + bF;
        float gg = acc[2][v] + bG;
        float go = acc[3][v] + bO;
        float si = 1.f / (1.f + __expf(-gi));
        float sf = 1.f / (1.f + __expf(-gf));
        float so = 1.f / (1.f + __expf(-go));
        float cn = sf * c[cm] + si * tanhf(gg);
        float hn = so * tanhf(cn);
        c[cm]   = cn;
        _Float16 hh = (_Float16)hn;
        act_write[aw + 512 + j]  = hh;            // h -> partner mid slot
        hStageOut[cm]            = hh;            // h -> own slot, next kernel
        act_write[aw + 1024 + j] = hStageIn[cm];  // partner's prev h -> self slot
        h32[cm] = hn;
        if (mode == 0) {
            act_write[aw + j] = act_read[aw + j];                 // copy x_t
        } else if (tNext < S_) {
            act_write[aw + j] = (_Float16)x[((size_t)m * S_ + tNext) * F_ + j];
        }
    }
}

// price_pred = ph @ pW.T + pb ; metrics_pred = mh @ mW.T + mb (fp32, tiny)
__global__ __launch_bounds__(256) void heads_kernel(
    const float* __restrict__ ph, const float* __restrict__ mh,
    const float* __restrict__ pW, const float* __restrict__ pb,
    const float* __restrict__ mW, const float* __restrict__ mb,
    float* __restrict__ out)
{
    int idx = blockIdx.x * 256 + threadIdx.x;      // 65536 threads
    int sel = idx >> 15;                            // 0 = price, 1 = metrics
    int loc = idx & 32767;
    int m = loc >> 6, n = loc & 63;
    const float* h  = sel ? mh : ph;
    const float* W  = sel ? mW : pW;
    const float* bb = sel ? mb : pb;
    float s = bb[n];
    const float* hr = h + (size_t)m * 512;
    const float* wr = W + (size_t)n * 512;
    #pragma unroll 8
    for (int k = 0; k < 512; ++k) s += hr[k] * wr[k];
    out[idx] = s;
}

extern "C" void kernel_launch(void* const* d_in, const int* in_sizes, int n_in,
                              void* d_out, int out_size, void* d_ws, size_t ws_size,
                              hipStream_t stream)
{
    const float* x    = (const float*)d_in[0];
    const float* mWih = (const float*)d_in[1];
    const float* mWhh = (const float*)d_in[2];
    const float* mbih = (const float*)d_in[3];
    const float* mbhh = (const float*)d_in[4];
    const float* pWih = (const float*)d_in[5];
    const float* pWhh = (const float*)d_in[6];
    const float* pbih = (const float*)d_in[7];
    const float* pbhh = (const float*)d_in[8];
    const float* pW   = (const float*)d_in[9];
    const float* pb   = (const float*)d_in[10];
    const float* mW   = (const float*)d_in[11];
    const float* mb   = (const float*)d_in[12];
    float* out = (float*)d_out;

    char* ws = (char*)d_ws;
    size_t off = 0;
    auto alloc = [&](size_t bytes) -> char* {
        char* p = ws + off;
        off += (bytes + 255) & ~(size_t)255;
        return p;
    };
    _Float16* BmP     = (_Float16*)alloc((size_t)KD * NG * 2); // 6 MB
    _Float16* BpP     = (_Float16*)alloc((size_t)KD * NG * 2); // 6 MB
    float*    biasM   = (float*)alloc(NG * 4);
    float*    biasP   = (float*)alloc(NG * 4);
    _Float16* actM    = (_Float16*)alloc((size_t)B_ * KD * 2); // 1.5 MB
    _Float16* actP    = (_Float16*)alloc((size_t)B_ * KD * 2);
    _Float16* mhStage = (_Float16*)alloc((size_t)B_ * E_ * 2); // 0.5 MB
    _Float16* phStage = (_Float16*)alloc((size_t)B_ * E_ * 2);
    float*    mc      = (float*)alloc((size_t)B_ * E_ * 4);
    float*    pc      = (float*)alloc((size_t)B_ * E_ * 4);
    float*    mh32    = (float*)alloc((size_t)B_ * E_ * 4);
    float*    ph32    = (float*)alloc((size_t)B_ * E_ * 4);

    // one-time (per launch) weight/bias packing + state init + x_0
    pack_b_kernel<<<(NNT * NKT * 32) / 256, 256, 0, stream>>>(mWih, mWhh, BmP);
    pack_b_kernel<<<(NNT * NKT * 32) / 256, 256, 0, stream>>>(pWih, pWhh, BpP);
    bias_init_kernel<<<16, 256, 0, stream>>>(mbih, mbhh, pbih, pbhh, biasM, biasP);
    init_state_kernel<<<(B_ * 1024) / 256, 256, 0, stream>>>(actM, phStage, mc, pc);
    convert_x_kernel<<<(B_ * F_) / 256, 256, 0, stream>>>(x, actM);

    dim3 grid(B_ / 16, 8);                          // 32 x 8 blocks of 4 waves
    for (int t = 0; t < S_; ++t) {
        // metrics: read actM=[x|ph|mh], write actP=[x|mh_new|ph(stage)]
        lstm_cell_kernel<<<grid, 128, 0, stream>>>(
            actM, BmP, biasM, mc, actP, mhStage, phStage, mh32,
            x, 0, /*mode=*/0);
        // price: read actP=[x|mh|ph], write actM=[x_{t+1}|ph_new|mh(stage)]
        lstm_cell_kernel<<<grid, 128, 0, stream>>>(
            actP, BpP, biasP, pc, actM, phStage, mhStage, ph32,
            x, t + 1, /*mode=*/1);
    }

    heads_kernel<<<(2 * B_ * 64) / 256, 256, 0, stream>>>(
        ph32, mh32, pW, pb, mW, mb, out);
}
